// AttenBlock_34119220199667
// MI455X (gfx1250) — matrix operand order repaired and checked by reading, not compiled, not run on hardware
//
#include <hip/hip_runtime.h>
#include <hip/hip_bf16.h>
#include <math.h>

// ---------------- problem constants ----------------
#define Bc      32
#define Lc      1021
#define Dc      512
#define KKc     1536      // conv GEMM K = 3*512
#define INTERc  2048
#define TWODc   1024      // 2*D
#define M2c     4096      // INPUT_LEN
#define LPc     1024      // padded L for token GEMM
#define LT32    32        // ceil(1021/32) L-tiles of 32

// ---------------- workspace layout (bytes) ----------------
#define OFF_XNH     ((size_t)0)                       // f16 [B,L,512]
#define SZ_ROWH     ((size_t)Bc*Lc*Dc*2)
#define OFF_LASTXH  (OFF_XNH + SZ_ROWH)
#define OFF_LXH     (OFF_LASTXH + SZ_ROWH)
#define OFF_FEATH   (OFF_LXH + SZ_ROWH)               // f16 [B,L,1024]
#define SZ_FEATH    ((size_t)Bc*Lc*TWODc*2)
#define OFF_OUT1T   (OFF_FEATH + SZ_FEATH)            // f16 [B,512,1024] (d-major, l padded)
#define SZ_OUT1T    ((size_t)Bc*Dc*LPc*2)
#define OFF_ATTEN   (OFF_OUT1T + SZ_OUT1T)            // f32 [B*L,4]
#define SZ_ATTEN    ((size_t)Bc*Lc*4*4)
#define OFF_CONVWT  (OFF_ATTEN + SZ_ATTEN)            // f16 [512,1536]  (o, k*512+i)
#define SZ_CONVWT   ((size_t)Dc*KKc*2)
#define OFF_GATEH   (OFF_CONVWT + SZ_CONVWT)          // f16 [2048,1024]
#define SZ_GATEH    ((size_t)INTERc*TWODc*2)
#define OFF_UPH     (OFF_GATEH + SZ_GATEH)
#define OFF_DOWNH   (OFF_UPH + SZ_GATEH)              // f16 [512,2048]
#define SZ_DOWNH    ((size_t)Dc*INTERc*2)
#define OFF_TOKWH   (OFF_DOWNH + SZ_DOWNH)            // f16 [4096,1024] zero-padded K
#define SZ_TOKWH    ((size_t)M2c*LPc*2)

// ---------------- WMMA helpers ----------------
typedef __attribute__((ext_vector_type(16))) _Float16 v16h;
typedef __attribute__((ext_vector_type(8)))  float    v8f;

union FragU { v16h h; int4 q[2]; };

__device__ inline v16h frag_from2(const _Float16* p0, const _Float16* p1) {
    FragU u;
    u.q[0] = *reinterpret_cast<const int4*>(p0);
    u.q[1] = *reinterpret_cast<const int4*>(p1);
    return u.h;
}
__device__ inline v16h frag_zero() {
    FragU u;
    u.q[0] = make_int4(0, 0, 0, 0);
    u.q[1] = make_int4(0, 0, 0, 0);
    return u.h;
}
__device__ inline v8f wmma_f16(v16h a, v16h b, v8f c) {
    return __builtin_amdgcn_wmma_f32_16x16x32_f16(false, a, false, b, (short)0, c, false, false);
}

// ---------------- weight conversion kernels ----------------
__global__ void k_cvt(const float* __restrict__ s, _Float16* __restrict__ d, int n) {
    int i = blockIdx.x * 256 + threadIdx.x;
    if (i < n) d[i] = (_Float16)s[i];
}

// conv_w [O=512, I=512, K=3] (OIH) -> convwt[o][k*512+i]
__global__ void k_convw(const float* __restrict__ s, _Float16* __restrict__ d) {
    int i = blockIdx.x * 256 + threadIdx.x;
    if (i < Dc * KKc) {
        int o = i / KKc, kk = i % KKc;
        int k = kk >> 9, c = kk & 511;
        d[i] = (_Float16)s[((o << 9) + c) * 3 + k];
    }
}

// token_w [4096,1021] -> f16 [4096,1024] zero padded
__global__ void k_tokw(const float* __restrict__ s, _Float16* __restrict__ d) {
    int i = blockIdx.x * 256 + threadIdx.x;
    if (i < M2c * LPc) {
        int m = i >> 10, l = i & 1023;
        d[i] = (l < Lc) ? (_Float16)s[(size_t)m * Lc + l] : (_Float16)0.0f;
    }
}

// ---------------- LayerNorm(x) -> xnh ; last_x -> lastxh ----------------
__global__ void __launch_bounds__(256) k_ln(const float* __restrict__ x,
                                            const float* __restrict__ lastx,
                                            const float* __restrict__ g,
                                            const float* __restrict__ be,
                                            _Float16* __restrict__ xnh,
                                            _Float16* __restrict__ lastxh) {
    size_t t = blockIdx.x;                 // token index in [0, B*L)
    const float* xr = x + t * Dc;
    int tid = threadIdx.x;
    float v0 = xr[tid], v1 = xr[tid + 256];
    float s = v0 + v1, s2 = v0 * v0 + v1 * v1;
    for (int off = 16; off > 0; off >>= 1) {
        s  += __shfl_xor(s,  off, 32);
        s2 += __shfl_xor(s2, off, 32);
    }
    __shared__ float red[16];
    int lane = tid & 31, w = tid >> 5;
    if (lane == 0) { red[w] = s; red[8 + w] = s2; }
    __syncthreads();
    if (tid == 0) {
        float a = 0.f, c = 0.f;
        for (int i = 0; i < 8; ++i) { a += red[i]; c += red[8 + i]; }
        red[0] = a; red[8] = c;
    }
    __syncthreads();
    float mean = red[0] * (1.0f / 512.0f);
    float var  = red[8] * (1.0f / 512.0f) - mean * mean;
    float rs = rsqrtf(var + 1e-5f);
    xnh[t * Dc + tid]       = (_Float16)((v0 - mean) * rs * g[tid]       + be[tid]);
    xnh[t * Dc + tid + 256] = (_Float16)((v1 - mean) * rs * g[tid + 256] + be[tid + 256]);
    lastxh[t * Dc + tid]       = (_Float16)lastx[t * Dc + tid];
    lastxh[t * Dc + tid + 256] = (_Float16)lastx[t * Dc + tid + 256];
}

// ---------------- conv1d as shifted WMMA GEMM (2x2 blocked): lxh = conv(last_x) ----------------
__global__ void __launch_bounds__(256) k_conv_gemm(const _Float16* __restrict__ lastxh,
                                                   const _Float16* __restrict__ convwt,
                                                   _Float16* __restrict__ lxh) {
    int w = threadIdx.x >> 5, lane = threadIdx.x & 31;
    int tid = blockIdx.x * 8 + w;               // wave macro-tile id
    int b = tid / (LT32 * 16);
    int rem = tid % (LT32 * 16);
    int lt = rem >> 4, nt = rem & 15;
    int l0 = lt * 32, o0 = nt * 32;
    int mA = lane & 15;
    int koff  = (lane < 16) ? 0 : 8;
    int koffB = (lane < 16) ? 0 : 8;
    int n = lane & 15, mbase = (lane < 16) ? 0 : 8;
    const _Float16* brow0 = convwt + (size_t)(o0 + n) * KKc;
    const _Float16* brow1 = convwt + (size_t)(o0 + 16 + n) * KKc;
    v8f c00 = {}, c01 = {}, c10 = {}, c11 = {};
    for (int kk = 0; kk < KKc; kk += 32) {
        int k1 = kk + koff;
        int seg = k1 >> 9;            // conv tap
        int i1 = k1 & 511;            // channel
        int tok0 = l0 + mA + seg - 1; // shifted source row (pad=1)
        int tok1 = tok0 + 16;
        v16h a0, a1;
        if (tok0 >= 0 && tok0 < Lc) {
            const _Float16* ar = lastxh + ((size_t)(b * Lc + tok0)) * Dc + i1;
            a0 = frag_from2(ar, ar + 16);
        } else a0 = frag_zero();
        if (tok1 < Lc) {
            const _Float16* ar = lastxh + ((size_t)(b * Lc + tok1)) * Dc + i1;
            a1 = frag_from2(ar, ar + 16);
        } else a1 = frag_zero();
        v16h b0 = frag_from2(brow0 + kk + koffB, brow0 + kk + koffB + 16);
        v16h b1 = frag_from2(brow1 + kk + koffB, brow1 + kk + koffB + 16);
        c00 = wmma_f16(a0, b0, c00);
        c01 = wmma_f16(a0, b1, c01);
        c10 = wmma_f16(a1, b0, c10);
        c11 = wmma_f16(a1, b1, c11);
    }
    for (int r = 0; r < 8; ++r) {
        int l = l0 + mbase + r;
        if (l < Lc) {
            _Float16* row = lxh + ((size_t)(b * Lc + l)) * Dc;
            row[o0 + n]      = (_Float16)c00[r];
            row[o0 + 16 + n] = (_Float16)c01[r];
        }
        int l2 = l + 16;
        if (l2 < Lc) {
            _Float16* row = lxh + ((size_t)(b * Lc + l2)) * Dc;
            row[o0 + n]      = (_Float16)c10[r];
            row[o0 + 16 + n] = (_Float16)c11[r];
        }
    }
}

// ---------------- attention logits + softmax (K=3) ----------------
__global__ void k_atten(const _Float16* __restrict__ lxh,
                        const float* __restrict__ aw,
                        float* __restrict__ atten) {
    int w = threadIdx.x >> 5, lane = threadIdx.x & 31;
    long t = (long)blockIdx.x * 8 + w;
    if (t >= (long)Bc * Lc) return;
    const _Float16* row = lxh + (size_t)t * Dc;
    float a0 = 0.f, a1 = 0.f, a2 = 0.f;
    for (int c = lane; c < Dc; c += 32) {
        float v = (float)row[c];
        a0 += v * aw[c];
        a1 += v * aw[Dc + c];
        a2 += v * aw[2 * Dc + c];
    }
    for (int off = 16; off > 0; off >>= 1) {
        a0 += __shfl_xor(a0, off, 32);
        a1 += __shfl_xor(a1, off, 32);
        a2 += __shfl_xor(a2, off, 32);
    }
    if (lane == 0) {
        float mx = fmaxf(a0, fmaxf(a1, a2));
        float e0 = __expf(a0 - mx), e1 = __expf(a1 - mx), e2 = __expf(a2 - mx);
        float inv = 1.0f / (e0 + e1 + e2);
        atten[t * 4 + 0] = e0 * inv;
        atten[t * 4 + 1] = e1 * inv;
        atten[t * 4 + 2] = e2 * inv;
        atten[t * 4 + 3] = 0.f;
    }
}

// ---------------- window mix -> feat [BL,1024] (coarse | fine) ----------------
__global__ void __launch_bounds__(256) k_feat(const _Float16* __restrict__ lxh,
                                              const _Float16* __restrict__ xnh,
                                              const float* __restrict__ atten,
                                              _Float16* __restrict__ feath) {
    size_t t = blockIdx.x;
    int l = (int)(t % Lc);
    float a0 = atten[t * 4], a1 = atten[t * 4 + 1], a2 = atten[t * 4 + 2];
    bool p0 = (l > 0), p2 = (l + 1 < Lc);
    size_t r0 = (t - 1) * Dc, r1 = t * Dc, r2 = (t + 1) * Dc;
    for (int j = threadIdx.x; j < TWODc; j += 256) {
        const _Float16* src = (j < Dc) ? lxh : xnh;
        int c = j & 511;
        float v = a1 * (float)src[r1 + c];
        if (p0) v += a0 * (float)src[r0 + c];
        if (p2) v += a2 * (float)src[r2 + c];
        feath[t * TWODc + j] = (_Float16)v;
    }
}

// ---------------- fused SwiGLU MLP (32 tokens/block, 2x2 blocked) ----------------
// out1t[b][d][l] = down(silu(feat@gate^T) * (feat@up^T)), stored d-major, l padded to 1024
__global__ void __launch_bounds__(256) k_mlp(const _Float16* __restrict__ feath,
                                             const _Float16* __restrict__ gateh,
                                             const _Float16* __restrict__ uph,
                                             const _Float16* __restrict__ downh,
                                             _Float16* __restrict__ out1t) {
    extern __shared__ _Float16 sH[];       // [32][2048] = 128 KB h tile
    int w = threadIdx.x >> 5, lane = threadIdx.x & 31;
    int b = blockIdx.x >> 5;       // /32
    int lt = blockIdx.x & 31;
    int l0 = lt * 32;
    int mA = lane & 15;
    int koff  = (lane < 16) ? 0 : 8;
    int koffB = (lane < 16) ? 0 : 8;
    int n = lane & 15, mbase = (lane < 16) ? 0 : 8;

    int tokA0 = l0 + mA, tokA1 = l0 + 16 + mA;
    bool aok0 = (tokA0 < Lc), aok1 = (tokA1 < Lc);
    const _Float16* arow0 = feath + ((size_t)(b * Lc + (aok0 ? tokA0 : 0))) * TWODc;
    const _Float16* arow1 = feath + ((size_t)(b * Lc + (aok1 ? tokA1 : 0))) * TWODc;

    // stage 1: gate/up GEMMs + SiLU combine -> sH (128 N-tiles, 8 waves x 16)
    for (int j = 0; j < 16; ++j) {
        int nt = w + 8 * j;
        const _Float16* grow = gateh + (size_t)(nt * 16 + n) * TWODc;
        const _Float16* urow = uph   + (size_t)(nt * 16 + n) * TWODc;
        v8f g0 = {}, g1 = {}, u0 = {}, u1 = {};
        for (int k = 0; k < TWODc; k += 32) {
            v16h a0 = aok0 ? frag_from2(arow0 + k + koff, arow0 + k + koff + 16) : frag_zero();
            v16h a1 = aok1 ? frag_from2(arow1 + k + koff, arow1 + k + koff + 16) : frag_zero();
            v16h bg = frag_from2(grow + k + koffB, grow + k + koffB + 16);
            v16h bu = frag_from2(urow + k + koffB, urow + k + koffB + 16);
            g0 = wmma_f16(a0, bg, g0);
            u0 = wmma_f16(a0, bu, u0);
            g1 = wmma_f16(a1, bg, g1);
            u1 = wmma_f16(a1, bu, u1);
        }
        for (int r = 0; r < 8; ++r) {
            float gv0 = g0[r];
            float hv0 = (gv0 / (1.0f + __expf(-gv0))) * u0[r];
            sH[(mbase + r) * INTERc + nt * 16 + n] = (_Float16)hv0;
            float gv1 = g1[r];
            float hv1 = (gv1 / (1.0f + __expf(-gv1))) * u1[r];
            sH[(16 + mbase + r) * INTERc + nt * 16 + n] = (_Float16)hv1;
        }
    }
    __syncthreads();

    // stage 2: down projection (32 N-tiles, 8 waves x 4), store transposed+padded
    const _Float16* ah0 = sH + (size_t)mA * INTERc;
    const _Float16* ah1 = sH + (size_t)(16 + mA) * INTERc;
    for (int j = 0; j < 4; ++j) {
        int nt = w * 4 + j;
        const _Float16* drow = downh + (size_t)(nt * 16 + n) * INTERc;
        v8f c0 = {}, c1 = {};
        for (int k = 0; k < INTERc; k += 32) {
            v16h a0 = frag_from2(ah0 + k + koff, ah0 + k + koff + 16);
            v16h a1 = frag_from2(ah1 + k + koff, ah1 + k + koff + 16);
            v16h bd = frag_from2(drow + k + koffB, drow + k + koffB + 16);
            c0 = wmma_f16(a0, bd, c0);
            c1 = wmma_f16(a1, bd, c1);
        }
        int d = nt * 16 + n;
        _Float16* orow = out1t + ((size_t)(b * Dc + d)) * LPc;
        for (int r = 0; r < 8; ++r) {
            orow[l0 + mbase + r]      = (_Float16)c0[r];
            orow[l0 + 16 + mbase + r] = (_Float16)c1[r];
        }
    }
}

// ---------------- token projection (2x2 blocked): out[b,m,d] = sum_l tokw[m,l]*out1[b,l,d] ----------------
__global__ void __launch_bounds__(256) k_token_gemm(const _Float16* __restrict__ tokwh,
                                                    const _Float16* __restrict__ out1t,
                                                    float* __restrict__ out) {
    int w = threadIdx.x >> 5, lane = threadIdx.x & 31;
    int tid = blockIdx.x * 8 + w;
    int b = tid / (128 * 16);
    int rem = tid % (128 * 16);
    int mt = rem >> 4, nt = rem & 15;
    int m0 = mt * 32, d0 = nt * 32;
    int mA = lane & 15;
    int koff  = (lane < 16) ? 0 : 8;
    int koffB = (lane < 16) ? 0 : 8;
    int n = lane & 15, mbase = (lane < 16) ? 0 : 8;
    const _Float16* arow0 = tokwh + (size_t)(m0 + mA) * LPc;
    const _Float16* arow1 = tokwh + (size_t)(m0 + 16 + mA) * LPc;
    const _Float16* brow0 = out1t + ((size_t)(b * Dc + d0 + n)) * LPc;
    const _Float16* brow1 = out1t + ((size_t)(b * Dc + d0 + 16 + n)) * LPc;
    v8f c00 = {}, c01 = {}, c10 = {}, c11 = {};
    for (int k = 0; k < LPc; k += 32) {
        v16h a0 = frag_from2(arow0 + k + koff, arow0 + k + koff + 16);
        v16h a1 = frag_from2(arow1 + k + koff, arow1 + k + koff + 16);
        v16h b0 = frag_from2(brow0 + k + koffB, brow0 + k + koffB + 16);
        v16h b1 = frag_from2(brow1 + k + koffB, brow1 + k + koffB + 16);
        c00 = wmma_f16(a0, b0, c00);
        c01 = wmma_f16(a0, b1, c01);
        c10 = wmma_f16(a1, b0, c10);
        c11 = wmma_f16(a1, b1, c11);
    }
    for (int r = 0; r < 8; ++r) {
        float* row0 = out + ((size_t)(b * M2c + m0 + mbase + r)) * Dc;
        row0[d0 + n]      = c00[r];
        row0[d0 + 16 + n] = c01[r];
        float* row1 = out + ((size_t)(b * M2c + m0 + 16 + mbase + r)) * Dc;
        row1[d0 + n]      = c10[r];
        row1[d0 + 16 + n] = c11[r];
    }
}

// ---------------- host launch ----------------
extern "C" void kernel_launch(void* const* d_in, const int* in_sizes, int n_in,
                              void* d_out, int out_size, void* d_ws, size_t ws_size,
                              hipStream_t stream) {
    const float* x      = (const float*)d_in[0];
    const float* last_x = (const float*)d_in[1];
    const float* ln_g   = (const float*)d_in[2];
    const float* ln_b   = (const float*)d_in[3];
    const float* conv_w = (const float*)d_in[4];
    const float* aw     = (const float*)d_in[5];
    const float* gate_w = (const float*)d_in[6];
    const float* up_w   = (const float*)d_in[7];
    const float* down_w = (const float*)d_in[8];
    const float* tok_w  = (const float*)d_in[9];
    float* out = (float*)d_out;

    char* ws = (char*)d_ws;
    _Float16* xnh    = (_Float16*)(ws + OFF_XNH);
    _Float16* lastxh = (_Float16*)(ws + OFF_LASTXH);
    _Float16* lxh    = (_Float16*)(ws + OFF_LXH);
    _Float16* feath  = (_Float16*)(ws + OFF_FEATH);
    _Float16* out1t  = (_Float16*)(ws + OFF_OUT1T);
    float*    atten  = (float*)   (ws + OFF_ATTEN);
    _Float16* convwt = (_Float16*)(ws + OFF_CONVWT);
    _Float16* gateh  = (_Float16*)(ws + OFF_GATEH);
    _Float16* uph    = (_Float16*)(ws + OFF_UPH);
    _Float16* downh  = (_Float16*)(ws + OFF_DOWNH);
    _Float16* tokwh  = (_Float16*)(ws + OFF_TOKWH);

    // weight conversions
    k_cvt  <<<(INTERc * TWODc + 255) / 256, 256, 0, stream>>>(gate_w, gateh, INTERc * TWODc);
    k_cvt  <<<(INTERc * TWODc + 255) / 256, 256, 0, stream>>>(up_w,   uph,   INTERc * TWODc);
    k_cvt  <<<(Dc * INTERc + 255) / 256, 256, 0, stream>>>(down_w, downh, Dc * INTERc);
    k_convw<<<(Dc * KKc + 255) / 256, 256, 0, stream>>>(conv_w, convwt);
    k_tokw <<<(M2c * LPc + 255) / 256, 256, 0, stream>>>(tok_w, tokwh);

    // layernorm + f16 staging
    k_ln<<<Bc * Lc, 256, 0, stream>>>(x, last_x, ln_g, ln_b, xnh, lastxh);

    // conv GEMM (32x32 macro-tiles): B * 32 * 16 waves / 8 waves per block
    k_conv_gemm<<<(Bc * LT32 * 16) / 8, 256, 0, stream>>>(lastxh, convwt, lxh);

    // softmax attention
    k_atten<<<(Bc * Lc + 7) / 8, 256, 0, stream>>>(lxh, aw, atten);

    // window mix
    k_feat<<<Bc * Lc, 256, 0, stream>>>(lxh, xnh, atten, feath);

    // fused SwiGLU MLP: 32 tokens/block, 128 KB dynamic LDS
    k_mlp<<<Bc * 32, 256, 32 * INTERc * sizeof(_Float16), stream>>>(feath, gateh, uph, downh, out1t);

    // token projection (32x32 macro-tiles): B * 128 * 16 waves / 8
    k_token_gemm<<<(Bc * 128 * 16) / 8, 256, 0, stream>>>(tokwh, out1t, out);

    (void)in_sizes; (void)n_in; (void)out_size; (void)ws_size;
}
